// DeepBackward_19275813224820
// MI455X (gfx1250) — matrix operand order repaired and checked by reading, not compiled, not run on hardware
//
#include <hip/hip_runtime.h>

typedef _Float16 v16h __attribute__((ext_vector_type(16)));
typedef _Float16 v8h  __attribute__((ext_vector_type(8)));
typedef float    v8f  __attribute__((ext_vector_type(8)));

#define HDIM 512
#define BN_EPS 1e-5f

// ---------------------------------------------------------------------------
// zero scratch accumulators + loss output
__global__ void zero_kernel(float* __restrict__ sums, int n, float* __restrict__ out) {
    int i = blockIdx.x * blockDim.x + threadIdx.x;
    if (i < n) sums[i] = 0.0f;
    if (i == 0) out[0] = 0.0f;
}

// ---------------------------------------------------------------------------
// Transpose 4 hidden 512x512 fp32 weight matrices into f16, N-major:
// Wt[mat][n][k] = W[mat][k][n].  mat 0,1 = y_Wh[0..1]; mat 2,3 = z_Wh[0..1].
__global__ void transpose_w_kernel(const float* __restrict__ Wy,
                                   const float* __restrict__ Wz,
                                   _Float16* __restrict__ Wt) {
    int idx = blockIdx.x * blockDim.x + threadIdx.x;   // 4*512*512 total
    int mat = idx >> 18;
    int rem = idx & ((1 << 18) - 1);
    int n = rem >> 9;
    int k = rem & (HDIM - 1);
    const float* W = (mat < 2) ? (Wy + (size_t)mat * HDIM * HDIM)
                               : (Wz + (size_t)(mat - 2) * HDIM * HDIM);
    Wt[(size_t)mat * HDIM * HDIM + (size_t)n * HDIM + k] =
        (_Float16)W[(size_t)k * HDIM + n];
}

// ---------------------------------------------------------------------------
// Batch stats of x and relu(x-1): writes {mu_x, rsig_x, mu_f, rsig_f}
__global__ __launch_bounds__(1024)
void input_stats_kernel(const float* __restrict__ x, int B, float* __restrict__ stats) {
    __shared__ float s0[1024], s1[1024], s2[1024], s3[1024];
    int tid = threadIdx.x;
    float sx = 0.f, sxx = 0.f, sf = 0.f, sff = 0.f;
    for (int i = tid; i < B; i += 1024) {
        float xv = x[i];
        float fv = fmaxf(xv - 1.0f, 0.0f);
        sx += xv; sxx += xv * xv; sf += fv; sff += fv * fv;
    }
    s0[tid] = sx; s1[tid] = sxx; s2[tid] = sf; s3[tid] = sff;
    __syncthreads();
    for (int s = 512; s > 0; s >>= 1) {
        if (tid < s) {
            s0[tid] += s0[tid + s]; s1[tid] += s1[tid + s];
            s2[tid] += s2[tid + s]; s3[tid] += s3[tid + s];
        }
        __syncthreads();
    }
    if (tid == 0) {
        float invB = 1.0f / (float)B;
        float mux = s0[0] * invB;
        float varx = s1[0] * invB - mux * mux;
        float muf = s2[0] * invB;
        float varf = s3[0] * invB - muf * muf;
        stats[0] = mux;
        stats[1] = rsqrtf(varx + BN_EPS);
        stats[2] = muf;
        stats[3] = rsqrtf(varf + BN_EPS);
    }
}

// ---------------------------------------------------------------------------
// Input layer: U[b][c] = BN0(input) @ W_in + b_in   (K = 1 or 2, elementwise)
__global__ __launch_bounds__(256)
void input_layer_kernel(const float* __restrict__ x,
                        const float* __restrict__ stats,
                        const float* __restrict__ g0, const float* __restrict__ b0,
                        const float* __restrict__ Win, const float* __restrict__ bin,
                        float* __restrict__ U, int isY) {
    int idx = blockIdx.x * blockDim.x + threadIdx.x;
    int b = idx >> 9;
    int c = idx & (HDIM - 1);
    float xv = x[b];
    float mux = stats[0], rsx = stats[1];
    float c0 = (xv - mux) * rsx * g0[0] + b0[0];
    float u;
    if (isY) {
        float fx = fmaxf(xv - 1.0f, 0.0f);
        float muf = stats[2], rsf = stats[3];
        float c1 = (fx - muf) * rsf * g0[1] + b0[1];
        u = c0 * Win[c] + c1 * Win[HDIM + c] + bin[c];
    } else {
        u = c0 * Win[c] + bin[c];
    }
    U[idx] = u;
}

// ---------------------------------------------------------------------------
// Column sums / sum-of-squares of U (B x 512).  One thread per column,
// 256 rows per block, float atomics into a zeroed accumulator.
__global__ __launch_bounds__(512)
void colstats_kernel(const float* __restrict__ U, float* __restrict__ sums) {
    int c = threadIdx.x;                  // 0..511
    int r0 = blockIdx.x * 256;
    const float* p = U + (size_t)r0 * HDIM + c;
    float s = 0.f, s2 = 0.f;
    #pragma unroll 8
    for (int r = 0; r < 256; ++r) {
        float v = p[(size_t)r * HDIM];
        s += v; s2 += v * v;
    }
    unsafeAtomicAdd(&sums[c], s);
    unsafeAtomicAdd(&sums[HDIM + c], s2);
}

// ---------------------------------------------------------------------------
// BN (training-mode batch stats) + ReLU + f16 convert
__global__ __launch_bounds__(256)
void bn_relu_kernel(const float* __restrict__ U, const float* __restrict__ sums,
                    const float* __restrict__ g, const float* __restrict__ bt,
                    _Float16* __restrict__ Hb, float invB) {
    int idx = blockIdx.x * blockDim.x + threadIdx.x;
    int c = idx & (HDIM - 1);
    float mean = sums[c] * invB;
    float var = sums[HDIM + c] * invB - mean * mean;
    float rs = rsqrtf(var + BN_EPS);
    float h = (U[idx] - mean) * rs * g[c] + bt[c];
    Hb[idx] = (_Float16)fmaxf(h, 0.0f);
}

// ---------------------------------------------------------------------------
__device__ __forceinline__ v16h packA(v8h lo, v8h hi) {
    v16h a;
    #pragma unroll
    for (int i = 0; i < 8; ++i) { a[i] = lo[i]; a[8 + i] = hi[i]; }
    return a;
}

// WMMA GEMM:  U[B x 512] = A[B x 512](f16) @ W[512 x 512] + bias
// W supplied transposed/N-major as Wt[n][k] (f16).
// Wave tile: 16(M) x 64(N), 4 accumulators; block = 8 waves = 32M x 256N.
// K-loop unrolled by 2 with ping-pong fragment sets (NO rotation copies, so
// the register allocator cannot coalesce prefetch targets onto live WMMA
// sources): loads of one set overlap with the 4 WMMAs of the other set.
// Fragment layouts as loaded here (wave32, contiguous K, same order for A and B):
//   A 16x32 f16 : lanes 0-15 row M=lane, halves0-7 = K 0..7, halves8-15 = K 8..15
//                 lanes16-31 same rows, K 16..23 / 24..31      -> two b128 loads
//   B 32x16 f16 : lanes 0-15 col N=lane, halves0-15 = K 0..15
//                 lanes16-31 col N=lane-16, K 16..31          -> one 32B load
//   C 16x16 f32 : lane<16: N=lane, vgpr r = row r ; lane>=16: row r+8
__global__ __launch_bounds__(256)
void gemm512_wmma_kernel(const _Float16* __restrict__ A,
                         const _Float16* __restrict__ Wt,
                         const float* __restrict__ bias,
                         float* __restrict__ U, int Brows) {
    const int K = HDIM, N = HDIM;
    int tid  = threadIdx.x;
    int lane = tid & 31;
    int wave = tid >> 5;            // 0..7
    int waveM = wave >> 2;          // 0..1
    int waveN = wave & 3;           // 0..3
    int bm = blockIdx.x >> 1;       // M block (32 rows)
    int bn = blockIdx.x & 1;        // N block (256 cols)
    int mBase = bm * 32 + waveM * 16;
    int nBase = bn * 256 + waveN * 64;

    int l15 = lane & 15;
    int hi  = lane >> 4;

    const _Float16* aPtr = A  + (size_t)(mBase + l15) * K + hi * 16;
    const _Float16* bPtr = Wt + (size_t)(nBase + l15) * K + hi * 16;

    // set0 @ kb = 0
    v8h  a0lo = *(const v8h*)(aPtr);
    v8h  a0hi = *(const v8h*)(aPtr + 8);
    v16h b00  = *(const v16h*)(bPtr);
    v16h b01  = *(const v16h*)(bPtr + (size_t)16 * K);
    v16h b02  = *(const v16h*)(bPtr + (size_t)32 * K);
    v16h b03  = *(const v16h*)(bPtr + (size_t)48 * K);

    v8f c0 = {}, c1 = {}, c2 = {}, c3 = {};

    #pragma unroll
    for (int kb = 0; kb < K; kb += 64) {
        // set1 @ kb+32 : issue before consuming set0
        v8h  a1lo = *(const v8h*)(aPtr + kb + 32);
        v8h  a1hi = *(const v8h*)(aPtr + kb + 40);
        v16h b10  = *(const v16h*)(bPtr + kb + 32);
        v16h b11  = *(const v16h*)(bPtr + (size_t)16 * K + kb + 32);
        v16h b12  = *(const v16h*)(bPtr + (size_t)32 * K + kb + 32);
        v16h b13  = *(const v16h*)(bPtr + (size_t)48 * K + kb + 32);

        {
            v16h a = packA(a0lo, a0hi);
            c0 = __builtin_amdgcn_wmma_f32_16x16x32_f16(false, a, false, b00, (short)0, c0, false, false);
            c1 = __builtin_amdgcn_wmma_f32_16x16x32_f16(false, a, false, b01, (short)0, c1, false, false);
            c2 = __builtin_amdgcn_wmma_f32_16x16x32_f16(false, a, false, b02, (short)0, c2, false, false);
            c3 = __builtin_amdgcn_wmma_f32_16x16x32_f16(false, a, false, b03, (short)0, c3, false, false);
        }

        // set0 @ kb+64 : issue before consuming set1 (skipped on last iter)
        if (kb + 64 < K) {
            a0lo = *(const v8h*)(aPtr + kb + 64);
            a0hi = *(const v8h*)(aPtr + kb + 72);
            b00  = *(const v16h*)(bPtr + kb + 64);
            b01  = *(const v16h*)(bPtr + (size_t)16 * K + kb + 64);
            b02  = *(const v16h*)(bPtr + (size_t)32 * K + kb + 64);
            b03  = *(const v16h*)(bPtr + (size_t)48 * K + kb + 64);
        }

        {
            v16h a = packA(a1lo, a1hi);
            c0 = __builtin_amdgcn_wmma_f32_16x16x32_f16(false, a, false, b10, (short)0, c0, false, false);
            c1 = __builtin_amdgcn_wmma_f32_16x16x32_f16(false, a, false, b11, (short)0, c1, false, false);
            c2 = __builtin_amdgcn_wmma_f32_16x16x32_f16(false, a, false, b12, (short)0, c2, false, false);
            c3 = __builtin_amdgcn_wmma_f32_16x16x32_f16(false, a, false, b13, (short)0, c3, false, false);
        }
    }

    int col = nBase + l15;
    int rowBase = mBase + hi * 8;
    float bb0 = bias[col], bb1 = bias[col + 16], bb2 = bias[col + 32], bb3 = bias[col + 48];
    #pragma unroll
    for (int r = 0; r < 8; ++r) {
        size_t ro = (size_t)(rowBase + r) * N + col;
        U[ro]      = c0[r] + bb0;
        U[ro + 16] = c1[r] + bb1;
        U[ro + 32] = c2[r] + bb2;
        U[ro + 48] = c3[r] + bb3;
    }
}

// ---------------------------------------------------------------------------
// Output layer: out[b] = dot(H[b][:], w) + b_out.  One wave32 per row.
__global__ __launch_bounds__(256)
void outdot_kernel(const _Float16* __restrict__ Hb, const float* __restrict__ w,
                   const float* __restrict__ bout, float* __restrict__ out, int Brows) {
    int wave = threadIdx.x >> 5, lane = threadIdx.x & 31;
    int row = blockIdx.x * 8 + wave;
    if (row >= Brows) return;
    const _Float16* hr = Hb + (size_t)row * HDIM;
    float s = 0.f;
    #pragma unroll 4
    for (int k = lane; k < HDIM; k += 32) s += (float)hr[k] * w[k];
    #pragma unroll
    for (int m = 16; m >= 1; m >>= 1) s += __shfl_xor(s, m, 32);
    if (lane == 0) out[row] = s + bout[0];
}

// ---------------------------------------------------------------------------
// Fused driver + MSE reduction
__global__ __launch_bounds__(256)
void loss_kernel(const float* __restrict__ x, const float* __restrict__ x_next,
                 const float* __restrict__ dw, const float* __restrict__ outY,
                 const float* __restrict__ outZ, int B, float* __restrict__ lossOut) {
    __shared__ float red[256];
    int tid = threadIdx.x;
    int b = blockIdx.x * blockDim.x + tid;
    float acc = 0.f;
    if (b < B) {
        float xv = x[b];
        float F = fmaxf(xv - 1.0f, 0.0f);
        float yt = outY[b] + F;            // y_NNs_forward adds F_x
        float zt = outZ[b];
        float delta = F - yt;              // K_CTRL = 1
        float a = -delta;                  // -(K/LAMBDA)*delta
        float sp = fmaxf(a, 0.0f) + log1pf(expf(-fabsf(a)));   // logaddexp(0,a)
        float f = -0.05f * yt - 0.1f * fabsf(zt) + delta + sp;
        float yn = fmaxf(x_next[b] - 1.0f, 0.0f);
        float t = yn - yt + f * 0.02f - zt * dw[b];
        acc = t * t;
    }
    red[tid] = acc;
    __syncthreads();
    for (int s = 128; s > 0; s >>= 1) {
        if (tid < s) red[tid] += red[tid + s];
        __syncthreads();
    }
    if (tid == 0) unsafeAtomicAdd(lossOut, red[0] / (float)B);
}

// ---------------------------------------------------------------------------
extern "C" void kernel_launch(void* const* d_in, const int* in_sizes, int n_in,
                              void* d_out, int out_size, void* d_ws, size_t ws_size,
                              hipStream_t stream) {
    const int Bn = in_sizes[0];     // 32768
    const int H = HDIM;

    const float* x        = (const float*)d_in[0];
    const float* x_next   = (const float*)d_in[1];
    const float* dw       = (const float*)d_in[2];
    const float* y_bn0_g  = (const float*)d_in[3];
    const float* y_bn0_b  = (const float*)d_in[4];
    const float* y_W_in   = (const float*)d_in[5];
    const float* y_b_in   = (const float*)d_in[6];
    const float* y_Wh     = (const float*)d_in[7];
    const float* y_bh     = (const float*)d_in[8];
    const float* y_bn_g   = (const float*)d_in[9];
    const float* y_bn_b   = (const float*)d_in[10];
    const float* y_W_out  = (const float*)d_in[11];
    const float* y_b_out  = (const float*)d_in[12];
    const float* z_bn0_g  = (const float*)d_in[13];
    const float* z_bn0_b  = (const float*)d_in[14];
    const float* z_W_in   = (const float*)d_in[15];
    const float* z_b_in   = (const float*)d_in[16];
    const float* z_Wh     = (const float*)d_in[17];
    const float* z_bh     = (const float*)d_in[18];
    const float* z_bn_g   = (const float*)d_in[19];
    const float* z_bn_b   = (const float*)d_in[20];
    const float* z_W_out  = (const float*)d_in[21];
    const float* z_b_out  = (const float*)d_in[22];

    // workspace layout
    char* ws = (char*)d_ws;
    float*    U     = (float*)ws;                                   // B*H fp32
    _Float16* Hb    = (_Float16*)(ws + (size_t)Bn * H * 4);         // B*H f16
    _Float16* Wt    = (_Float16*)(ws + (size_t)Bn * H * 6);         // 4*H*H f16
    float*    sums  = (float*)(ws + (size_t)Bn * H * 6 + (size_t)4 * H * H * 2); // 6*2*H
    float*    stats = sums + 6 * 2 * H;                             // 4 floats
    float*    yt    = stats + 16;                                   // B fp32
    float*    zt    = yt + Bn;                                      // B fp32
    float*    lossOut = (float*)d_out;

    const float invB = 1.0f / (float)Bn;
    const int elems = Bn * H;                // B*H elementwise threads

    // 1) zero accumulators + output
    zero_kernel<<<(6 * 2 * H + 255) / 256, 256, 0, stream>>>(sums, 6 * 2 * H, lossOut);

    // 2) transpose hidden weights -> f16 N-major
    transpose_w_kernel<<<(4 * H * H) / 256, 256, 0, stream>>>(y_Wh, z_Wh, Wt);

    // 3) input column stats (x, relu(x-1))
    input_stats_kernel<<<1, 1024, 0, stream>>>(x, Bn, stats);

    // 4) per-net pipeline (net 0 = y, net 1 = z), buffers reused
    const float* bn0g[2] = { y_bn0_g, z_bn0_g };
    const float* bn0b[2] = { y_bn0_b, z_bn0_b };
    const float* Win[2]  = { y_W_in, z_W_in };
    const float* bin[2]  = { y_b_in, z_b_in };
    const float* bh[2]   = { y_bh, z_bh };
    const float* bng[2]  = { y_bn_g, z_bn_g };
    const float* bnb[2]  = { y_bn_b, z_bn_b };
    const float* Wout[2] = { y_W_out, z_W_out };
    const float* bout[2] = { y_b_out, z_b_out };
    float* netOut[2]     = { yt, zt };

    for (int net = 0; net < 2; ++net) {
        float* s0 = sums + (size_t)(net * 3 + 0) * 2 * H;
        float* s1 = sums + (size_t)(net * 3 + 1) * 2 * H;
        float* s2 = sums + (size_t)(net * 3 + 2) * 2 * H;

        input_layer_kernel<<<elems / 256, 256, 0, stream>>>(
            x, stats, bn0g[net], bn0b[net], Win[net], bin[net], U, net == 0 ? 1 : 0);

        colstats_kernel<<<Bn / 256, 512, 0, stream>>>(U, s0);
        bn_relu_kernel<<<elems / 256, 256, 0, stream>>>(
            U, s0, bng[net] + 0 * H, bnb[net] + 0 * H, Hb, invB);

        // hidden layer 0
        gemm512_wmma_kernel<<<(Bn / 32) * 2, 256, 0, stream>>>(
            Hb, Wt + (size_t)(net * 2 + 0) * H * H, bh[net] + 0 * H, U, Bn);
        colstats_kernel<<<Bn / 256, 512, 0, stream>>>(U, s1);
        bn_relu_kernel<<<elems / 256, 256, 0, stream>>>(
            U, s1, bng[net] + 1 * H, bnb[net] + 1 * H, Hb, invB);

        // hidden layer 1
        gemm512_wmma_kernel<<<(Bn / 32) * 2, 256, 0, stream>>>(
            Hb, Wt + (size_t)(net * 2 + 1) * H * H, bh[net] + 1 * H, U, Bn);
        colstats_kernel<<<Bn / 256, 512, 0, stream>>>(U, s2);
        bn_relu_kernel<<<elems / 256, 256, 0, stream>>>(
            U, s2, bng[net] + 2 * H, bnb[net] + 2 * H, Hb, invB);

        // output layer 512 -> 1
        outdot_kernel<<<Bn / 8, 256, 0, stream>>>(Hb, Wout[net], bout[net], netOut[net], Bn);
    }

    // 5) fused driver + MSE
    loss_kernel<<<Bn / 256, 256, 0, stream>>>(x, x_next, dw, yt, zt, Bn, lossOut);
}
